// DownSampler_7017976561791
// MI455X (gfx1250) — matrix operand order repaired and checked by reading, not compiled, not run on hardware
//
#include <hip/hip_runtime.h>

// ---------------------------------------------------------------------------
// Fully fused 3-stage downsampler for MI455X (gfx1250, wave32, WMMA).
// One workgroup = one 16x16 tile of the final 128x128 output of one image.
// All intermediates live in LDS (~117 KB of the 320 KB WGP pool).
// Stage 1 & 2 run on v_wmma_f32_16x16x32_bf16; stage 3 is f32 VALU (b128 LDS).
// Store-back is branch-free: tiles are padded to whole 16-row M-tiles and
// invalid channels are redirected (cndmask on address) to an LDS dump region.
// ---------------------------------------------------------------------------

typedef __attribute__((ext_vector_type(16))) __bf16 v16bf;
typedef __attribute__((ext_vector_type(8)))  float  v8f;

union BFrag { unsigned int u[8]; v16bf v; };

__device__ __forceinline__ unsigned short f2bf(float f) {
    unsigned int u = __float_as_uint(f);
    unsigned int r = u + 0x7FFFu + ((u >> 16) & 1u);   // round-to-nearest-even
    return (unsigned short)(r >> 16);
}

// reflect-pad-1 index maps (pad coord -> source coord), clamped for don't-care halos
__device__ __forceinline__ int refl258(int p) {        // stage-1 output grid (256)
    p = p < 0 ? 0 : (p > 257 ? 257 : p);
    return p == 0 ? 1 : (p == 257 ? 254 : p - 1);
}
__device__ __forceinline__ int refl514(int q) {        // input grid (512)
    q = q < 0 ? 0 : (q > 513 ? 513 : q);
    return q == 0 ? 1 : (q == 513 ? 510 : q - 1);
}

// LDS byte offsets
#define OFF_A1     0          // 1456*20 bf16 = 58240  (38x38 pixels padded to 91 M-tiles)
#define OFF_W2T    58240      // 32*320  bf16 = 20480
#define OFF_X      78720      // 78*80   bf16 = 12480
#define OFF_W1T    91200      // 32*16   bf16 =  1024
#define OFF_A2     92224      // 336*20  f32  = 26880  (18x18 pixels padded to 21 M-tiles)
#define OFF_W3T    119104     // 9*20    f32  =   720  (transposed: [k][ci])
#define OFF_DUMPH  119824     // 32 bf16 dump  =    64
#define OFF_DUMPF  119888     // 32 f32  dump  =   128
#define LDS_BYTES  120016

__global__ __launch_bounds__(256) void downsampler_fused(
    const float* __restrict__ X, const float* __restrict__ W1,
    const float* __restrict__ W2, const float* __restrict__ W3,
    float* __restrict__ OUT)
{
    extern __shared__ char lds[];
    unsigned short* sA1  = (unsigned short*)(lds + OFF_A1);
    unsigned short* sW2T = (unsigned short*)(lds + OFF_W2T);
    unsigned short* sX   = (unsigned short*)(lds + OFF_X);
    unsigned short* sW1T = (unsigned short*)(lds + OFF_W1T);
    float*          sA2  = (float*)(lds + OFF_A2);
    float*          sW3T = (float*)(lds + OFF_W3T);

    const unsigned int* sA1u  = (const unsigned int*)sA1;
    const unsigned int* sW2Tu = (const unsigned int*)sW2T;
    const unsigned int* sXu   = (const unsigned int*)sX;
    const unsigned int* sW1Tu = (const unsigned int*)sW1T;

    const int DUMPH = (OFF_DUMPH - OFF_A1) / 2;   // ushort index rel. sA1
    const int DUMPF = (OFF_DUMPF - OFF_A2) / 4;   // float  index rel. sA2

    const int tid  = threadIdx.x;
    const int lane = tid & 31;          // wave32
    const int wave = tid >> 5;          // 8 waves
    const int hi   = lane >> 4;         // half-wave select (A/B lane layout)
    const int ln16 = lane & 15;

    const int tx = blockIdx.x;          // 0..7  (16-col output tile)
    const int ty = blockIdx.y;          // 0..7  (16-row output tile)
    const int n  = blockIdx.z;          // 0..47 (folded B*C image)

    const int p1r_base = 32 * ty - 2;   // stage-1 padded-coord window base (38 rows)
    const int p1c_base = 32 * tx - 2;
    const int qr_base  = 64 * ty - 6;   // input padded-coord window base (78 rows)
    const int qc_base  = 64 * tx - 6;

    // Warm L2 for the (tiny, all-blocks-shared) weight tensors.
    __builtin_prefetch(W1, 0, 2);
    __builtin_prefetch(W2, 0, 2);
    __builtin_prefetch(W3, 0, 2);

    // ---- Phase 0: weights -> LDS (bf16, channel-padded to 32, K-contiguous)
    for (int i = tid; i < 32 * 16; i += 256) {          // sW1T[co][kpos]
        int co = i >> 4, k = i & 15;
        sW1T[i] = f2bf(co < 20 ? W1[co * 16 + k] : 0.f);
    }
    for (int i = tid; i < 32 * 320; i += 256) {         // sW2T[co][K], K=(kr,kc,ci)
        int co = i / 320, K = i % 320;
        int kpos = K / 20, ci = K % 20;
        sW2T[i] = f2bf(co < 20 ? W2[(co * 20 + ci) * 16 + kpos] : 0.f);
    }
    for (int i = tid; i < 180; i += 256) {              // sW3T[k][ci] (transposed)
        int k = i / 20, ci = i % 20;
        sW3T[i] = W3[ci * 9 + k];
    }

    // ---- Phase A: reflect-padded 78x78 input patch -> LDS bf16 ------------
    const float* Xn = X + (size_t)n * 512 * 512;
    for (int i = tid; i < 78 * 78; i += 256) {
        int lr = i / 78, lc = i % 78;
        int xr = refl514(qr_base + lr);
        int xc = refl514(qc_base + lc);
        sX[lr * 80 + lc] = f2bf(Xn[xr * 512 + xc]);
    }
    __syncthreads();

    // ---- Phase B: stage-1 conv as WMMA. M=38x38 pixels (91 tiles), N=20(->2),
    //      K=16 patch taps zero-padded to 32. Weight B-fragments hoisted. ----
    BFrag b1[2];
    #pragma unroll
    for (int nt = 0; nt < 2; ++nt)
        #pragma unroll
        for (int j = 0; j < 8; ++j) {
            int co = nt * 16 + ln16;
            // B layout: VGPR j<4 -> K=hi*8+2j,+1 (taps 0..15 split by half),
            // VGPRs 4..7 hold K 16..31 which are the zero-padded taps.
            b1[nt].u[j] = (j < 4 && co < 20) ? sW1Tu[co * 8 + hi * 4 + j] : 0u;
        }

    for (int mt = wave; mt < 91; mt += 8) {
        BFrag a;
        int pi = mt * 16 + ln16;                       // <= 1455, tile padded
        int lpr = pi / 38, lpc = pi % 38;
        int r = refl258(p1r_base + lpr);               // stage-1 source row
        int c = refl258(p1c_base + lpc);
        #pragma unroll
        for (int v = 0; v < 4; ++v) {
            int kpos = hi * 8 + 2 * v;                 // A layout: K pairs per VGPR
            int kr = kpos >> 2, kc = kpos & 3;
            int lqr = 2 * r + kr - qr_base;
            int lqc = 2 * c + kc - qc_base;            // even -> aligned u32 pair
            a.u[v] = sXu[lqr * 40 + (lqc >> 1)];
            a.u[v + 4] = 0u;                           // K=16..31 zero padding
        }
        v8f acc0 = {}, acc1 = {};
        acc0 = __builtin_amdgcn_wmma_f32_16x16x32_bf16(false, a.v, false, b1[0].v,
                                                       (short)0, acc0, false, false);
        acc1 = __builtin_amdgcn_wmma_f32_16x16x32_bf16(false, a.v, false, b1[1].v,
                                                       (short)0, acc1, false, false);
        #pragma unroll
        for (int d = 0; d < 8; ++d) {                  // D: VGPR d -> M = d + hi*8
            int p = mt * 16 + d + hi * 8;              // always < 1456 (padded tile)
            float v0 = acc0[d]; v0 = v0 > 0.f ? v0 : 0.f;
            sA1[p * 20 + ln16] = f2bf(v0);             // relu, bf16 for stage 2
            int a1 = (ln16 < 4) ? (p * 20 + 16 + ln16) : (DUMPH + lane);
            float v1 = acc1[d]; v1 = v1 > 0.f ? v1 : 0.f;
            sA1[a1] = f2bf(v1);                        // branch-free redirect
        }
    }
    __syncthreads();

    // ---- Phase C: stage-2 conv as WMMA. M=18x18 pixels (21 tiles), N=20(->2),
    //      K=320=(kr,kc,ci) in 10 chunks of 32. k outermost: B built once per
    //      chunk per wave; up to 3 persistent M-tile accumulators per wave. --
    {
        v8f acc[3][2];
        int prer[3], prec[3];
        #pragma unroll
        for (int t = 0; t < 3; ++t) {
            acc[t][0] = (v8f){}; acc[t][1] = (v8f){};
            int mt = wave + 8 * t;
            int pi = (mt < 21 ? mt : 20) * 16 + ln16;
            prer[t] = pi / 18; prec[t] = pi % 18;
        }
        #pragma unroll
        for (int k = 0; k < 10; ++k) {
            BFrag b0, b1f;
            #pragma unroll
            for (int j = 0; j < 8; ++j) {
                int K = k * 32 + ((j < 4) ? (2 * j) : (2 * j + 8)) + hi * 8;
                b0.u[j]  = sW2Tu[ln16 * 160 + (K >> 1)];
                b1f.u[j] = (ln16 < 4) ? sW2Tu[(16 + ln16) * 160 + (K >> 1)] : 0u;
            }
            #pragma unroll
            for (int t = 0; t < 3; ++t) {
                int mt = wave + 8 * t;
                if (mt >= 21) break;                   // wave-uniform
                BFrag a;
                #pragma unroll
                for (int v = 0; v < 8; ++v) {
                    int kk = ((v < 4) ? (2 * v) : (2 * v + 8)) + hi * 8;
                    int K  = k * 32 + kk;
                    int kpos = K / 20, ci = K % 20;    // K even -> ci even
                    int kr = kpos >> 2, kc = kpos & 3;
                    int row = 2 * prer[t] + kr, col = 2 * prec[t] + kc;
                    a.u[v] = sA1u[((row * 38 + col) * 20 + ci) >> 1];
                }
                acc[t][0] = __builtin_amdgcn_wmma_f32_16x16x32_bf16(false, a.v, false, b0.v,
                                                                    (short)0, acc[t][0], false, false);
                acc[t][1] = __builtin_amdgcn_wmma_f32_16x16x32_bf16(false, a.v, false, b1f.v,
                                                                    (short)0, acc[t][1], false, false);
            }
        }
        #pragma unroll
        for (int t = 0; t < 3; ++t) {
            int mt = wave + 8 * t;
            if (mt >= 21) break;                       // wave-uniform
            #pragma unroll
            for (int d = 0; d < 8; ++d) {
                int p = mt * 16 + d + hi * 8;          // always < 336 (padded tile)
                int r2 = p / 18, c2 = p % 18;
                int g2r = 16 * ty - 1 + r2, g2c = 16 * tx - 1 + c2;
                bool ok = (g2r >= 0) & (g2r < 128) & (g2c >= 0) & (g2c < 128);
                float v0 = acc[t][0][d];
                v0 = (ok && v0 > 0.f) ? v0 : 0.f;      // relu + zero-pad halo
                sA2[(r2 * 18 + c2) * 20 + ln16] = v0;
                int a1 = (ln16 < 4) ? ((r2 * 18 + c2) * 20 + 16 + ln16) : (DUMPF + lane);
                float v1 = acc[t][1][d];
                v1 = (ok && v1 > 0.f) ? v1 : 0.f;
                sA2[a1] = v1;                          // branch-free redirect
            }
        }
    }
    __syncthreads();

    // ---- Phase D: stage-3 3x3x20 conv + relu, one output pixel per thread --
    {
        int a3r = tid >> 4, a3c = tid & 15;
        float acc = 0.f;
        #pragma unroll
        for (int kr = 0; kr < 3; ++kr)
            #pragma unroll
            for (int kc = 0; kc < 3; ++kc) {
                const float4* av = (const float4*)&sA2[((a3r + kr) * 18 + (a3c + kc)) * 20];
                const float4* wv = (const float4*)&sW3T[(kr * 3 + kc) * 20];
                #pragma unroll
                for (int q = 0; q < 5; ++q) {          // b128 LDS loads
                    float4 a4 = av[q];
                    float4 w4 = wv[q];
                    acc += a4.x * w4.x + a4.y * w4.y + a4.z * w4.z + a4.w * w4.w;
                }
            }
        acc = acc > 0.f ? acc : 0.f;
        OUT[((size_t)n * 128 + (16 * ty + a3r)) * 128 + (16 * tx + a3c)] = acc;
    }
}

extern "C" void kernel_launch(void* const* d_in, const int* in_sizes, int n_in,
                              void* d_out, int out_size, void* d_ws, size_t ws_size,
                              hipStream_t stream) {
    (void)in_sizes; (void)n_in; (void)out_size; (void)d_ws; (void)ws_size;
    const float* x  = (const float*)d_in[0];   // [16,3,512,512] -> [48,512,512]
    const float* w1 = (const float*)d_in[1];   // [20,1,4,4]
    const float* w2 = (const float*)d_in[2];   // [20,20,4,4]
    const float* w3 = (const float*)d_in[3];   // [1,20,3,3]
    float* out = (float*)d_out;                // [48,128,128]
    dim3 grid(8, 8, 48), block(256);
    downsampler_fused<<<grid, block, LDS_BYTES, stream>>>(x, w1, w2, w3, out);
}
